// AsymetricKernel_71270687310186
// MI455X (gfx1250) — compile-verified, matrix operand order checked by reading, not run on hardware
//
#include <hip/hip_runtime.h>
#include <hip/hip_bf16.h>

// ---------------------------------------------------------------------------
// Linear attention (B=4, N=8192, DIM=512, H=8, DH=64) for gfx1250 (MI455X).
//   1) convert u_x / Wq / Wk / Wv to bf16
//   2) QKV GEMM (bf16 WMMA): W head-slice staged to LDS in *fragment order*
//      via async global->LDS (global_load_async_to_lds_b128, ASYNCcnt);
//      fused epilogue (instnorm / 2D rotary) writes Qr [b,h,n,64],
//      Ktr/Vtr [b,h,64,n] bf16.
//   3) dots = k^T v per (b,h): split-K over 8 segments, f32 atomics, then
//      convert to bf16 dotsT [b,h,d,c].
//   4) out = (q . dots)/N -> fp32 [b,n,512] (WMMA, K=64).
// ---------------------------------------------------------------------------

typedef __bf16 bf16;
typedef __attribute__((ext_vector_type(16))) __bf16 v16bf;
typedef __attribute__((ext_vector_type(8)))  float  v8f;
typedef int v4i __attribute__((vector_size(16)));   // matches builtin param

#define B_    4
#define N_    8192
#define DIM_  512
#define H_    8
#define DH_   64
#define BN_   (B_ * N_)
#define DSEG_ 8                     // split-K segments for k^T v
#define SEGLEN_ (N_ / DSEG_)

#if defined(__has_builtin)
#  if __has_builtin(__builtin_amdgcn_global_load_async_to_lds_b128)
#    define HAVE_ASYNC_LDS 1
#  endif
#endif
#ifndef HAVE_ASYNC_LDS
#  define HAVE_ASYNC_LDS 0
#endif

typedef __attribute__((address_space(1))) v4i* gv4i_p;   // __device__ int4*
typedef __attribute__((address_space(3))) v4i* lv4i_p;   // __shared__ int4*

static __device__ __forceinline__ v8f wmma_bf16(v16bf a, v16bf b, v8f c) {
  // (neg_a, A, neg_b, B, c_mod, C, reuse_a, reuse_b)
  return __builtin_amdgcn_wmma_f32_16x16x32_bf16(false, a, false, b,
                                                 (short)0, c, false, false);
}

// ---------------------------------------------------------------- convert ---
__global__ void f32_to_bf16(const float* __restrict__ src,
                            bf16* __restrict__ dst, int n) {
  int i = blockIdx.x * blockDim.x + threadIdx.x;
  int stride = gridDim.x * blockDim.x;
  for (; i < n; i += stride) dst[i] = (bf16)src[i];
}

__global__ void zero_f32(float* __restrict__ dst, int n) {
  int i = blockIdx.x * blockDim.x + threadIdx.x;
  int stride = gridDim.x * blockDim.x;
  for (; i < n; i += stride) dst[i] = 0.0f;
}

// ------------------------------------------------- QKV GEMM + fused epilog ---
// grid = (BN/128, H); block = 256 (8 waves). Wave: 16 tokens x 64 cols (1 head).
// mode: 0 = q (rotary), 1 = k (instnorm + rotary), 2 = v (instnorm)
//
// W head-slice (64 rows x 512 K = 64KB bf16) is staged into LDS in B-fragment
// order: chunk cid = ((kstep*4 + sub)*32 + lane) holds the 16 bf16 the given
// lane feeds to WMMA for (kstep, sub). Inner-loop LDS reads are then fully
// sequential across the wave (bank-conflict free).
__global__ __launch_bounds__(256) void qkv_gemm_fused(
    const bf16* __restrict__ X,    // [BN, DIM] bf16 row-major
    const bf16* __restrict__ W,    // [DIM, DIM] bf16 row-major [e,c]
    const float* __restrict__ pos, // [N, 2] fp32
    bf16* __restrict__ out,        // mode0: [B,H,N,64]; else [B,H,64,N]
    int mode) {
  const int lane = threadIdx.x & 31;
  const int wave = threadIdx.x >> 5;
  const int half = lane >> 4;
  const int l16  = lane & 15;
  const int rowBase = blockIdx.x * 128 + wave * 16;
  const int h = blockIdx.y;
  const int colBase = h * DH_;

  __shared__ bf16 sW[DH_ * DIM_];              // 64 KB, fragment-ordered

  // ---- stage W[colBase..colBase+63, :] into LDS, permuted to fragment order
  {
    const bf16* gW = W + (size_t)colBase * DIM_;
    #pragma unroll
    for (int it = 0; it < 8; ++it) {
      int cid  = it * 256 + (int)threadIdx.x;  // 2048 chunks of 32B
      int step = cid >> 7;                     // K-step (K = step*32)
      int sub  = (cid >> 5) & 3;               // 16-col subtile
      int ln   = cid & 31;                     // destination lane
      size_t srcOff = (size_t)(sub * 16 + (ln & 15)) * DIM_ +
                      step * 32 + (ln >> 4) * 16;   // elements
      const char* src = (const char*)gW + srcOff * 2;
      char*       dst = (char*)sW + (size_t)cid * 32;
#if HAVE_ASYNC_LDS
      __builtin_amdgcn_global_load_async_to_lds_b128(
          (gv4i_p)src, (lv4i_p)dst, 0, 0);
      __builtin_amdgcn_global_load_async_to_lds_b128(
          (gv4i_p)(src + 16), (lv4i_p)(dst + 16), 0, 0);
#else
      ((v16bf*)sW)[cid] = *(const v16bf*)(gW + srcOff);
#endif
    }
#if HAVE_ASYNC_LDS
#  if defined(__has_builtin) && __has_builtin(__builtin_amdgcn_s_wait_asynccnt)
    __builtin_amdgcn_s_wait_asynccnt(0);
#  else
    asm volatile("s_wait_asynccnt 0" ::: "memory");
#  endif
#endif
    __syncthreads();
  }

  v8f acc[4] = {v8f{}, v8f{}, v8f{}, v8f{}};

  // A: lane holds row (rowBase+l16), 16 contiguous K starting at half*16.
  const bf16* aRow = X + (size_t)(rowBase + l16) * DIM_ + half * 16;
  const v16bf* sB = (const v16bf*)sW;

  for (int s = 0; s < DIM_ / 32; ++s) {
    v16bf a  = *(const v16bf*)(aRow + s * 32);
    v16bf b0 = sB[(s * 4 + 0) * 32 + lane];
    v16bf b1 = sB[(s * 4 + 1) * 32 + lane];
    v16bf b2 = sB[(s * 4 + 2) * 32 + lane];
    v16bf b3 = sB[(s * 4 + 3) * 32 + lane];
    if (s + 1 < DIM_ / 32)
      __builtin_prefetch(aRow + s * 32 + 32, 0, 3);   // global_prefetch_b8
    acc[0] = wmma_bf16(a, b0, acc[0]);
    acc[1] = wmma_bf16(a, b1, acc[1]);
    acc[2] = wmma_bf16(a, b2, acc[2]);
    acc[3] = wmma_bf16(a, b3, acc[3]);
  }

  // ---- InstanceNorm over the 64-wide head dim (k and v). Row M = r+8*half
  // lives entirely in one 16-lane half across the 4 accumulators.
  if (mode != 0) {
    for (int r = 0; r < 8; ++r) {
      float v0 = acc[0][r], v1 = acc[1][r], v2 = acc[2][r], v3 = acc[3][r];
      float s  = v0 + v1 + v2 + v3;
      float s2 = v0 * v0 + v1 * v1 + v2 * v2 + v3 * v3;
      #pragma unroll
      for (int m = 8; m >= 1; m >>= 1) {      // stays within 16-lane half
        s  += __shfl_xor(s,  m, 32);
        s2 += __shfl_xor(s2, m, 32);
      }
      float mu  = s * (1.0f / 64.0f);
      float var = s2 * (1.0f / 64.0f) - mu * mu;
      float rs  = rsqrtf(var + 1e-5f);
      acc[0][r] = (v0 - mu) * rs;
      acc[1][r] = (v1 - mu) * rs;
      acc[2][r] = (v2 - mu) * rs;
      acc[3][r] = (v3 - mu) * rs;
    }
  }

  // ---- 2D rotary (q and k). cols 0..31 use pos_x, 32..63 use pos_y.
  // Partner element (c +/- 16) is the neighbouring accumulator, same lane/r.
  if (mode != 2) {
    // inv_freq[jj] = 10000^(-jj/16), jj = l16
    float inv_freq = __expf(-(float)l16 * (9.210340371976184f / 16.0f));
    for (int r = 0; r < 8; ++r) {
      int row = rowBase + r + 8 * half;
      int n   = row & (N_ - 1);
      float fx = pos[2 * n]     * 64.0f * inv_freq;
      float fy = pos[2 * n + 1] * 64.0f * inv_freq;
      float cx = __cosf(fx), sx = __sinf(fx);
      float cy = __cosf(fy), sy = __sinf(fy);
      float t0 = acc[0][r], t1 = acc[1][r];
      acc[0][r] = t0 * cx - t1 * sx;
      acc[1][r] = t1 * cx + t0 * sx;
      float u0 = acc[2][r], u1 = acc[3][r];
      acc[2][r] = u0 * cy - u1 * sy;
      acc[3][r] = u1 * cy + u0 * sy;
    }
  }

  // ---- store bf16: q row-major [b,h,n,64]; k/v transposed [b,h,64,n]
  for (int sub = 0; sub < 4; ++sub) {
    for (int r = 0; r < 8; ++r) {
      int row = rowBase + r + 8 * half;
      int n   = row & (N_ - 1);
      int bb  = row >> 13;                 // N = 2^13
      int c   = sub * 16 + l16;
      size_t idx;
      if (mode == 0)
        idx = ((size_t)(bb * H_ + h) * N_ + n) * DH_ + c;
      else
        idx = ((size_t)(bb * H_ + h) * DH_ + c) * N_ + n;
      out[idx] = (bf16)acc[sub][r];
    }
  }
}

// ------------------------------------------- dots partials (split-K atomics) ---
// grid = (B*H, DSEG). Wave w owns tiles 2w, 2w+1; accumulates SEGLEN of K,
// then global_atomic_add_f32 into dotsF[b*h][d][c].
__global__ __launch_bounds__(256) void kv_dots_partial(
    const bf16* __restrict__ Kt,   // [B*H, 64, N]
    const bf16* __restrict__ Vt,   // [B*H, 64, N]
    float* __restrict__ dotsF) {   // [B*H, 64(d), 64(c)] fp32, pre-zeroed
  const int lane = threadIdx.x & 31;
  const int wave = threadIdx.x >> 5;
  const int half = lane >> 4;
  const int l16  = lane & 15;
  const int bh   = blockIdx.x;
  const int seg  = blockIdx.y;
  const int t0   = 2 * wave;
  const int cT   = t0 >> 2;
  const int dT0  = t0 & 3, dT1 = dT0 + 1;

  v8f acc0 = {}, acc1 = {};
  const size_t segBase = (size_t)seg * SEGLEN_ + half * 16;
  const bf16* kb  = Kt + ((size_t)bh * DH_ + cT  * 16 + l16) * N_ + segBase;
  const bf16* vb0 = Vt + ((size_t)bh * DH_ + dT0 * 16 + l16) * N_ + segBase;
  const bf16* vb1 = Vt + ((size_t)bh * DH_ + dT1 * 16 + l16) * N_ + segBase;

  for (int i0 = 0; i0 < SEGLEN_; i0 += 32) {
    v16bf a  = *(const v16bf*)(kb  + i0);
    v16bf b0 = *(const v16bf*)(vb0 + i0);
    v16bf b1 = *(const v16bf*)(vb1 + i0);
    if (i0 + 32 < SEGLEN_) __builtin_prefetch(kb + i0 + 32, 0, 3);
    acc0 = wmma_bf16(a, b0, acc0);
    acc1 = wmma_bf16(a, b1, acc1);
  }

  for (int r = 0; r < 8; ++r) {
    int c  = cT * 16 + r + 8 * half;
    int d0 = dT0 * 16 + l16;
    int d1 = dT1 * 16 + l16;
    atomicAdd(&dotsF[(size_t)bh * 4096 + (size_t)d0 * 64 + c], acc0[r]);
    atomicAdd(&dotsF[(size_t)bh * 4096 + (size_t)d1 * 64 + c], acc1[r]);
  }
}

__global__ void dots_to_bf16(const float* __restrict__ src,
                             bf16* __restrict__ dst, int n) {
  int i = blockIdx.x * blockDim.x + threadIdx.x;
  int stride = gridDim.x * blockDim.x;
  for (; i < n; i += stride) dst[i] = (bf16)src[i];
}

// ---------------------------------------------------- out = (q . dots)/N ---
// grid = (N/128, B*H); wave: 16 tokens x 64 cols, K = 64 (2 WMMA steps).
__global__ __launch_bounds__(256) void q_dots_out(
    const bf16* __restrict__ Qr,     // [B*H, N, 64]
    const bf16* __restrict__ dotsT,  // [B*H, 64(d), 64(c)]
    float* __restrict__ out) {       // [B, N, 512]
  const int lane = threadIdx.x & 31;
  const int wave = threadIdx.x >> 5;
  const int half = lane >> 4;
  const int l16  = lane & 15;
  const int bh   = blockIdx.y;
  const int rowBase = blockIdx.x * 128 + wave * 16;

  v8f acc[4] = {v8f{}, v8f{}, v8f{}, v8f{}};
  const bf16* qb = Qr + ((size_t)bh * N_ + rowBase + l16) * DH_ + half * 16;
  const bf16* db = dotsT + (size_t)bh * 4096 + (size_t)l16 * 64 + half * 16;

  #pragma unroll
  for (int k0 = 0; k0 < DH_; k0 += 32) {
    v16bf a = *(const v16bf*)(qb + k0);
    #pragma unroll
    for (int sub = 0; sub < 4; ++sub) {
      v16bf b = *(const v16bf*)(db + (size_t)sub * 16 * 64 + k0);
      acc[sub] = wmma_bf16(a, b, acc[sub]);
    }
  }

  const int b = bh >> 3, h = bh & 7;
  const float scale = 1.0f / (float)N_;
  for (int sub = 0; sub < 4; ++sub) {
    for (int r = 0; r < 8; ++r) {
      int i = rowBase + r + 8 * half;
      int d = sub * 16 + l16;
      out[((size_t)(b * N_ + i)) * DIM_ + h * DH_ + d] = acc[sub][r] * scale;
    }
  }
}

// ------------------------------------------------------------------ launch ---
extern "C" void kernel_launch(void* const* d_in, const int* in_sizes, int n_in,
                              void* d_out, int out_size, void* d_ws,
                              size_t ws_size, hipStream_t stream) {
  (void)in_sizes; (void)n_in; (void)out_size; (void)ws_size;
  const float* u_x = (const float*)d_in[0];
  const float* pos = (const float*)d_in[1];
  const float* Wq  = (const float*)d_in[2];
  const float* Wk  = (const float*)d_in[3];
  const float* Wv  = (const float*)d_in[4];
  float* outp = (float*)d_out;

  char* ws = (char*)d_ws;
  size_t off = 0;
  bf16* Xb  = (bf16*)(ws + off); off += (size_t)BN_ * DIM_ * 2;       // 32 MB
  bf16* Wqb = (bf16*)(ws + off); off += (size_t)DIM_ * DIM_ * 2;
  bf16* Wkb = (bf16*)(ws + off); off += (size_t)DIM_ * DIM_ * 2;
  bf16* Wvb = (bf16*)(ws + off); off += (size_t)DIM_ * DIM_ * 2;
  bf16* Qr  = (bf16*)(ws + off); off += (size_t)B_ * H_ * N_ * DH_ * 2;
  bf16* Ktr = (bf16*)(ws + off); off += (size_t)B_ * H_ * N_ * DH_ * 2;
  bf16* Vtr = (bf16*)(ws + off); off += (size_t)B_ * H_ * N_ * DH_ * 2;
  bf16* dT  = (bf16*)(ws + off); off += (size_t)B_ * H_ * DH_ * DH_ * 2;
  float* dF = (float*)(ws + off); off += (size_t)B_ * H_ * DH_ * DH_ * 4;

  const int nDots = B_ * H_ * DH_ * DH_;      // 131072

  f32_to_bf16<<<4096, 256, 0, stream>>>(u_x, Xb, BN_ * DIM_);
  f32_to_bf16<<<512, 256, 0, stream>>>(Wq, Wqb, DIM_ * DIM_);
  f32_to_bf16<<<512, 256, 0, stream>>>(Wk, Wkb, DIM_ * DIM_);
  f32_to_bf16<<<512, 256, 0, stream>>>(Wv, Wvb, DIM_ * DIM_);
  zero_f32<<<128, 256, 0, stream>>>(dF, nDots);

  dim3 g1(BN_ / 128, H_);
  qkv_gemm_fused<<<g1, 256, 0, stream>>>(Xb, Wqb, pos, Qr, 0);
  qkv_gemm_fused<<<g1, 256, 0, stream>>>(Xb, Wkb, pos, Ktr, 1);
  qkv_gemm_fused<<<g1, 256, 0, stream>>>(Xb, Wvb, pos, Vtr, 2);

  dim3 g2(B_ * H_, DSEG_);
  kv_dots_partial<<<g2, 256, 0, stream>>>(Ktr, Vtr, dF);
  dots_to_bf16<<<128, 256, 0, stream>>>(dF, dT, nDots);

  dim3 g3(N_ / 128, B_ * H_);
  q_dots_out<<<g3, 256, 0, stream>>>(Qr, dT, outp);
}
